// RecurrentConvBlock_2422361555817
// MI455X (gfx1250) — matrix operand order repaired and checked by reading, not compiled, not run on hardware
//
#include <hip/hip_runtime.h>
#include <hip/hip_bf16.h>
#include <math.h>

// ---------------------------------------------------------------------------
// RecurrentConvBlock (4x ConvLSTM, residual, relu) for MI455X / gfx1250.
//
// Each ConvLSTM step = one implicit GEMM:
//   M = B*H*W pixels, N = 128 gate channels, K = 9*(Cin + F)
// run as v_wmma_f32_16x16x32_f16 (f16 in, f32 accumulate).
// t-major schedule (layer0..3 inside each timestep) => no sequence buffers,
// ~100MB of state stays resident in the 192MB L2.
// Block = 256 threads (8 wave32) = 32 pixels x 128 channels:
//   wave w owns N-tile w and TWO M-tiles (B-fragment reused by 2 WMMAs).
// ---------------------------------------------------------------------------

typedef __attribute__((ext_vector_type(16))) _Float16 v16h;
typedef __attribute__((ext_vector_type(4)))  _Float16 v4h;
typedef __attribute__((ext_vector_type(8)))  float    v8f;

#define BB 4
#define HH 128
#define WW 128
#define FF 32
#define NCH 128                 // 4*FF gate channels
#define MT 32                   // pixels per block (2 M-tiles)
#define KS_MAX (18*32 + 8)      // padded K stride, layers 1-3

// ---------------------------------------------------------------------------
// Pack weights (f32 [3,3,Cin,128] ++ [3,3,32,128]) -> f16 B-fragments:
// per (ntile, kchunk, lane) a contiguous 32B v16h.  B-fragment layout
// (16x16x32 f16, wave32): lane L -> column N = L%16; K = kc*32 + (L/16)*8 + i (i<8), kc*32 + 16 + (L/16)*8 + (i-8) (i>=8).
// ---------------------------------------------------------------------------
__global__ __launch_bounds__(256) void pack_weights_kernel(
    const float* __restrict__ Wx, const float* __restrict__ Wh,
    int Cx, int KC, _Float16* __restrict__ wp)
{
    const int Kx   = 9 * Cx;
    const int Ktot = Kx + 9 * FF;
    const int total = 8 * KC * 512;
    for (int idx = blockIdx.x * blockDim.x + threadIdx.x; idx < total;
         idx += gridDim.x * blockDim.x) {
        const int i    = idx & 15;
        const int lane = (idx >> 4) & 31;
        const int kc   = (idx >> 9) % KC;
        const int nt   = idx / (KC * 512);
        const int n    = nt * 16 + (lane & 15);
        const int k    = kc * 32 + (i < 8 ? (lane >> 4) * 8 + i : 16 + (lane >> 4) * 8 + (i - 8));
        float v = 0.0f;
        if (k < Kx)            v = Wx[k * NCH + n];        // [tap][c][n] flat
        else if (k < Ktot)     v = Wh[(k - Kx) * NCH + n];
        wp[idx] = (_Float16)v;
    }
}

// ---------------------------------------------------------------------------
// Fused ConvLSTM timestep.  MODE: 0 identity, 1 relu(in0), 2 relu(in0+in1)
// (transform applies to the layer input / x-part only).
// ---------------------------------------------------------------------------
template<int CX, int KC, int MODE>
__global__ __launch_bounds__(256) void convlstm_step_kernel(
    const float* __restrict__ in0, long in0_bstride, long in0_toff,
    const float* __restrict__ in1,
    const float* __restrict__ hprev,
    float*       __restrict__ cstate,
    float*       __restrict__ hnew,
    const _Float16* __restrict__ wp_raw,
    const float* __restrict__ bias)
{
    constexpr int KS   = KC * 32 + 8;     // padded K stride (16B align + bank spread)
    constexpr int Kx   = 9 * CX;
    constexpr int Ktot = Kx + 9 * FF;

    __shared__ _Float16 ldsA[MT * KS_MAX];   // im2col K-panel, [m][k]
    __shared__ float    zbuf[MT * NCH];      // gate pre-activations

    const int pix0 = blockIdx.x * MT;
    const int wv    = threadIdx.x >> 5;      // wave id == N-tile id (0..7)
    const int lane  = threadIdx.x & 31;
    const int g     = lane >> 4;
    const int mlane = lane & 15;

    const v16h* wp = (const v16h*)wp_raw;

    // Warm the wave's weight stream into L0/L2 while the A-panel is built.
#pragma unroll
    for (int kc = 0; kc < KC; ++kc)
        __builtin_prefetch(&wp[(wv * KC + kc) * 32 + lane], 0, 3);

    // ---- Stage A-panel: vectorized im2col (float4 -> 4 f16) ---------------
    // All tap/part/pad boundaries are multiples of 4 halfs, so each 4-run
    // stays inside one (source, tap) segment.
    constexpr int KS4 = KS / 4;
    for (int idx4 = threadIdx.x; idx4 < MT * KS4; idx4 += 256) {
        const int m  = idx4 / KS4;
        const int k4 = (idx4 - m * KS4) * 4;
        v4h hv = (v4h)0.0f;
        if (k4 < Ktot) {
            int tap, c;
            const float* src; long bstr, toff; int C;
            if (k4 < Kx) { tap = k4 / CX; c = k4 - tap * CX; C = CX;
                           src = in0; bstr = in0_bstride; toff = in0_toff; }
            else         { const int kk = k4 - Kx; tap = kk >> 5; c = kk & 31;
                           C = FF; src = hprev; bstr = (long)HH * WW * FF; toff = 0; }
            const int pix = pix0 + m;
            const int b  = pix >> 14;
            const int hw = pix & 16383;
            const int yy = (hw >> 7) + tap / 3 - 1;
            const int xx = (hw & 127) + tap % 3 - 1;
            if (yy >= 0 && yy < HH && xx >= 0 && xx < WW) {
                const long sp = (long)(yy * WW + xx);
                const float4 vv = *(const float4*)(src + (long)b * bstr + toff + sp * C + c);
                float4 w = vv;
                if (k4 < Kx) {
                    if (MODE == 1) {
                        w.x = fmaxf(w.x, 0.f); w.y = fmaxf(w.y, 0.f);
                        w.z = fmaxf(w.z, 0.f); w.w = fmaxf(w.w, 0.f);
                    } else if (MODE == 2) {
                        const float4 r = *(const float4*)(in1 + (long)b * ((long)HH * WW * FF) + sp * FF + c);
                        w.x = fmaxf(w.x + r.x, 0.f); w.y = fmaxf(w.y + r.y, 0.f);
                        w.z = fmaxf(w.z + r.z, 0.f); w.w = fmaxf(w.w + r.w, 0.f);
                    }
                }
                hv.x = (_Float16)w.x; hv.y = (_Float16)w.y;
                hv.z = (_Float16)w.z; hv.w = (_Float16)w.w;
            }
        }
        *(v4h*)(ldsA + m * KS + k4) = hv;
    }
    __syncthreads();

    // ---- WMMA K-loop: 2 M-tiles per wave, B-fragment reused ---------------
    const _Float16* arow0 = ldsA + mlane * KS;          // pixels 0..15
    const _Float16* arow1 = ldsA + (16 + mlane) * KS;   // pixels 16..31

    const float bv = bias[wv * 16 + mlane];             // fold bias into C
    v8f acc0 = {bv, bv, bv, bv, bv, bv, bv, bv};
    v8f acc1 = acc0;

#pragma unroll
    for (int kc = 0; kc < KC; ++kc) {
        // A-fragment (16x32 f16): lane holds row M = lane%16, group g = lane/16.
        // halfs 0..7 -> K = g*8 + 0..7 ; halfs 8..15 -> K = 16 + g*8 + 0..7
        // => two contiguous 16B LDS reads (ds_load_b128).
        const int koff = kc * 32 + g * 8;
        union { v16h v; uint4 q[2]; } a0, a1;
        a0.q[0] = *(const uint4*)(arow0 + koff);
        a0.q[1] = *(const uint4*)(arow0 + koff + 16);
        a1.q[0] = *(const uint4*)(arow1 + koff);
        a1.q[1] = *(const uint4*)(arow1 + koff + 16);
        const v16h bf = wp[(wv * KC + kc) * 32 + lane];
        acc0 = __builtin_amdgcn_wmma_f32_16x16x32_f16(
                   false, a0.v, false, bf, (short)0, acc0, false, false);
        acc1 = __builtin_amdgcn_wmma_f32_16x16x32_f16(
                   false, a1.v, false, bf, (short)0, acc1, false, false);
    }

    // ---- Gate pre-activations -> LDS --------------------------------------
    // C/D layout: lane holds N = lane%16; VGPR r holds M = g*8 + r.
#pragma unroll
    for (int r = 0; r < 8; ++r) {
        zbuf[(g * 8 + r) * NCH + wv * 16 + mlane]        = acc0[r];
        zbuf[(16 + g * 8 + r) * NCH + wv * 16 + mlane]   = acc1[r];
    }
    __syncthreads();

    // ---- Gates + state update (i,f,g,o channel blocks of 32) --------------
    for (int e = threadIdx.x; e < MT * FF; e += 256) {
        const int m = e >> 5;
        const int f = e & 31;
        const float zi = zbuf[m * NCH +       f];
        const float zf = zbuf[m * NCH +  32 + f];
        const float zg = zbuf[m * NCH +  64 + f];
        const float zo = zbuf[m * NCH +  96 + f];
        const float ig = fminf(fmaxf(0.2f * zi + 0.5f, 0.0f), 1.0f);
        const float fg = fminf(fmaxf(0.2f * zf + 0.5f, 0.0f), 1.0f);
        const float gg = tanhf(zg);
        const float og = fminf(fmaxf(0.2f * zo + 0.5f, 0.0f), 1.0f);
        const long off = (long)(pix0 + m) * FF + f;
        const float cn = fg * cstate[off] + ig * gg;
        cstate[off] = cn;
        hnew[off]   = og * tanhf(cn);
    }
}

// ---------------------------------------------------------------------------
// Host launcher.
//   ws (floats): h0[2], h1[2], h2[2], h3b, c0..c3 (11*S) then packed weights.
//   S = B*H*W*F = 2,097,152 floats (8.39MB) => ~93MB total workspace.
// ---------------------------------------------------------------------------
extern "C" void kernel_launch(void* const* d_in, const int* in_sizes, int n_in,
                              void* d_out, int out_size, void* d_ws, size_t ws_size,
                              hipStream_t stream) {
    (void)in_sizes; (void)n_in; (void)out_size; (void)ws_size;

    const float* x = (const float*)d_in[0];
    const float* Wx[4] = {(const float*)d_in[1], (const float*)d_in[4],
                          (const float*)d_in[7], (const float*)d_in[10]};
    const float* Wh[4] = {(const float*)d_in[2], (const float*)d_in[5],
                          (const float*)d_in[8], (const float*)d_in[11]};
    const float* bs[4] = {(const float*)d_in[3], (const float*)d_in[6],
                          (const float*)d_in[9], (const float*)d_in[12]};
    float* out = (float*)d_out;
    float* wsf = (float*)d_ws;

    const long S = (long)BB * HH * WW * FF;          // 2,097,152
    float* h0[2] = {wsf + 0 * S, wsf + 1 * S};
    float* h1[2] = {wsf + 2 * S, wsf + 3 * S};
    float* h2[2] = {wsf + 4 * S, wsf + 5 * S};
    float* h3[2] = {out, wsf + 6 * S};               // d_out doubles as h3 even buf
    float* cst[4] = {wsf + 7 * S, wsf + 8 * S, wsf + 9 * S, wsf + 10 * S};

    const int KC0 = 14, KC1 = 18;
    _Float16* wp0 = (_Float16*)(wsf + 11 * S);
    _Float16* wp1 = wp0 + (size_t)8 * KC0 * 512;
    _Float16* wp2 = wp1 + (size_t)8 * KC1 * 512;
    _Float16* wp3 = wp2 + (size_t)8 * KC1 * 512;

    // Zero the t=0 read-side state (capture-safe memset nodes).
    hipMemsetAsync(h0[0], 0, S * sizeof(float), stream);
    hipMemsetAsync(h1[0], 0, S * sizeof(float), stream);
    hipMemsetAsync(h2[0], 0, S * sizeof(float), stream);
    hipMemsetAsync(h3[0], 0, S * sizeof(float), stream);
    for (int l = 0; l < 4; ++l)
        hipMemsetAsync(cst[l], 0, S * sizeof(float), stream);

    // Pack weights into WMMA B-fragment order (recomputed each call).
    pack_weights_kernel<<<96, 256, 0, stream>>>(Wx[0], Wh[0], 16, KC0, wp0);
    pack_weights_kernel<<<96, 256, 0, stream>>>(Wx[1], Wh[1], 32, KC1, wp1);
    pack_weights_kernel<<<96, 256, 0, stream>>>(Wx[2], Wh[2], 32, KC1, wp2);
    pack_weights_kernel<<<96, 256, 0, stream>>>(Wx[3], Wh[3], 32, KC1, wp3);

    const int nblocks = (BB * HH * WW) / MT;         // 2048
    const long xbstride = (long)8 * HH * WW * 16;    // T*H*W*Cin
    const long sbstride = (long)HH * WW * FF;

    for (int t = 0; t < 8; ++t) {
        const int rb = t & 1, wb = 1 - rb;
        // convlstm0: input x[:,t]
        convlstm_step_kernel<16, 14, 0><<<nblocks, 256, 0, stream>>>(
            x, xbstride, (long)t * HH * WW * 16, nullptr,
            h0[rb], cst[0], h0[wb], wp0, bs[0]);
        // convlstm1: input h0(t)
        convlstm_step_kernel<32, 18, 0><<<nblocks, 256, 0, stream>>>(
            h0[wb], sbstride, 0, nullptr,
            h1[rb], cst[1], h1[wb], wp1, bs[1]);
        // convlstm2: input relu(h1(t))
        convlstm_step_kernel<32, 18, 1><<<nblocks, 256, 0, stream>>>(
            h1[wb], sbstride, 0, nullptr,
            h2[rb], cst[2], h2[wb], wp2, bs[2]);
        // convlstm3: input relu(h2(t) + y_c(t)),  y_c(t) = h0(t)
        convlstm_step_kernel<32, 18, 2><<<nblocks, 256, 0, stream>>>(
            h2[wb], sbstride, 0, h0[wb],
            h3[rb], cst[3], h3[wb], wp3, bs[3]);
    }
    // t=7 writes h3[0] == d_out.
}
